// GraphEnergyModel_45715631899093
// MI455X (gfx1250) — matrix operand order repaired and checked by reading, not compiled, not run on hardware
//
#include <hip/hip_runtime.h>

#define N_NODES 512
#define N_REL   4096
#define IN_DIM  4096
#define D_DIM   128
#define POOL_DIM 1024
#define NOBJ 151
#define NRELL 51
#define NN (N_NODES * N_NODES)

// ---------------- CDNA5 WMMA types ----------------
typedef __attribute__((ext_vector_type(16))) __bf16        v16bf;
typedef __attribute__((ext_vector_type(8)))  float         v8f;
typedef __attribute__((ext_vector_type(8)))  unsigned int  u32x8;

__device__ __forceinline__ unsigned short f2bf(float f) {
  unsigned int u = __float_as_uint(f);
  unsigned int r = u + 0x7FFFu + ((u >> 16) & 1u);
  return (unsigned short)(r >> 16);
}
__device__ __forceinline__ float sigmoidf(float x) { return 1.f / (1.f + __expf(-x)); }

// ---------------- WMMA GEMM: C(MxN=128) = act(A(MxK) @ B(Kx128) + bias) ----------------
// A: row-major bf16 (as u16). Bp: pre-packed bf16 fragments, layout:
//   Bp[(kc*8+nt)*512 + lane*16 + e] = B[kc*32 + 16*(lane>>4) + e, nt*16 + (lane&15)]
// NTW = N-tiles per wave; blockIdx.y selects the n-group (8/NTW groups).
// act: 0=none 1=relu 2=sigmoid
template <int NTW>
__global__ __launch_bounds__(256) void k_wmma_gemm_n128(
    const unsigned short* __restrict__ A, const unsigned short* __restrict__ Bp,
    const float* __restrict__ bias, float* __restrict__ C, int M, int K, int act)
{
  const int wave = threadIdx.x >> 5;
  const int lane = threadIdx.x & 31;
  const int mt = blockIdx.x * 8 + wave;
  const int m0 = mt << 4;
  if (m0 >= M) return;
  const int ng = blockIdx.y;  // n-group
  const int lh = lane >> 4;
  const int ln = lane & 15;
  const unsigned short* arow = A + (size_t)(m0 + ln) * K;

  const v8f vzero = {0.f, 0.f, 0.f, 0.f, 0.f, 0.f, 0.f, 0.f};
  v8f acc[NTW];
#pragma unroll
  for (int i = 0; i < NTW; ++i) acc[i] = vzero;

  const int nkc = K >> 5;
  for (int kc = 0; kc < nkc; ++kc) {
    // A fragment: per lane 16 contiguous bf16 at k = 16*lh + e (matches packed B)
    const unsigned int* p0 = (const unsigned int*)(arow + kc * 32 + 16 * lh);
    const unsigned int* p1 = (const unsigned int*)(arow + kc * 32 + 16 * lh + 8);
    u32x8 au;
    au[0] = p0[0]; au[1] = p0[1]; au[2] = p0[2]; au[3] = p0[3];
    au[4] = p1[0]; au[5] = p1[1]; au[6] = p1[2]; au[7] = p1[3];
    v16bf a = __builtin_bit_cast(v16bf, au);
#pragma unroll
    for (int j = 0; j < NTW; ++j) {
      const int nt = ng * NTW + j;
      const unsigned int* q =
          (const unsigned int*)(Bp + ((size_t)(kc * 8 + nt) << 9) + (lane << 4));
      u32x8 bu;
      bu[0] = q[0]; bu[1] = q[1]; bu[2] = q[2]; bu[3] = q[3];
      bu[4] = q[4]; bu[5] = q[5]; bu[6] = q[6]; bu[7] = q[7];
      v16bf b = __builtin_bit_cast(v16bf, bu);
      acc[j] = __builtin_amdgcn_wmma_f32_16x16x32_bf16(
          false, a, false, b, (short)0, acc[j], false, false);
    }
  }
#pragma unroll
  for (int j = 0; j < NTW; ++j) {
    const int col = ((ng * NTW + j) << 4) + ln;
    const float bs = bias ? bias[col] : 0.f;
#pragma unroll
    for (int g = 0; g < 8; ++g) {
      float v = acc[j][g] + bs;
      if (act == 1) v = fmaxf(v, 0.f);
      else if (act == 2) v = sigmoidf(v);
      C[(size_t)(m0 + (lh << 3) + g) * 128 + col] = v;  // D layout: M = 8*lh + g, N = lane&15
    }
  }
}

// ---------------- conversions / packing ----------------
__global__ void k_f2bf(const float* __restrict__ in, unsigned short* __restrict__ out, int n) {
  int i = blockIdx.x * blockDim.x + threadIdx.x;
  if (i < n) out[i] = f2bf(in[i]);
}

__global__ void k_packB(const float* __restrict__ B, unsigned short* __restrict__ out, int K) {
  int tid = blockIdx.x * blockDim.x + threadIdx.x;
  if (tid >= K * 128) return;
  int frag = tid >> 9, wi = tid & 511;
  int kc = frag >> 3, nt = frag & 7;
  int lane = wi >> 4, e = wi & 15;
  int lh = lane >> 4, n = lane & 15;
  int k = kc * 32 + lh * 16 + e;
  out[tid] = f2bf(B[k * 128 + nt * 16 + n]);
}

// ---------------- small dense stages ----------------
__global__ void k_pos1(const float* __restrict__ box, const float* __restrict__ w,
                       const float* __restrict__ b, const float* __restrict__ bng,
                       const float* __restrict__ bnb, float* __restrict__ t1) {
  int tid = blockIdx.x * blockDim.x + threadIdx.x;
  if (tid >= N_NODES * 32) return;
  int i = tid >> 5, c = tid & 31;
  float s = b[c];
  for (int k = 0; k < 9; ++k) s += box[i * 9 + k] * w[k * 32 + c];
  const float BN_SCALE = 0.99999500003749969f;  // 1/sqrt(1+1e-5)
  t1[tid] = s * BN_SCALE * bng[c] + bnb[c];
}

__global__ void k_pos2(const float* __restrict__ t1, const float* __restrict__ w,
                       const float* __restrict__ b, float* __restrict__ pos) {
  int tid = blockIdx.x * blockDim.x + threadIdx.x;
  if (tid >= N_NODES * D_DIM) return;
  int i = tid >> 7, c = tid & 127;
  float s = b[c];
  for (int k = 0; k < 32; ++k) s += t1[i * 32 + k] * w[k * 128 + c];
  pos[tid] = fmaxf(s, 0.f);
}

__global__ void k_sgh0(const float* __restrict__ objl, const float* __restrict__ pos,
                       const float* __restrict__ w, const float* __restrict__ b,
                       float* __restrict__ out) {
  int tid = blockIdx.x * blockDim.x + threadIdx.x;
  if (tid >= N_NODES * D_DIM) return;
  int i = tid >> 7, c = tid & 127;
  float s = b[c];
  for (int k = 0; k < NOBJ; ++k) s += objl[i * NOBJ + k] * w[k * 128 + c];
  for (int k = 0; k < 128; ++k) s += pos[i * 128 + k] * w[(NOBJ + k) * 128 + c];
  out[tid] = s;
}

__global__ void k_sge(const float* __restrict__ rell, const float* __restrict__ w,
                      const float* __restrict__ b, float* __restrict__ out) {
  int tid = blockIdx.x * blockDim.x + threadIdx.x;
  if (tid >= N_REL * D_DIM) return;
  int r = tid >> 7, c = tid & 127;
  float s = b[c];
  for (int k = 0; k < NRELL; ++k) s += rell[r * NRELL + k] * w[k * 128 + c];
  out[tid] = s;
}

// ---------------- sparse edge machinery ----------------
__global__ void k_mark(const int* __restrict__ rpi, int* __restrict__ mark) {
  int r = blockIdx.x * blockDim.x + threadIdx.x;
  if (r >= N_REL) return;
  mark[rpi[2 * r] * N_NODES + rpi[2 * r + 1]] = 1;
}

__global__ void k_compact(const int* __restrict__ mark, int* __restrict__ slotmap,
                          int* __restrict__ ekeys, int* __restrict__ cnt) {
  int key = blockIdx.x * blockDim.x + threadIdx.x;
  if (key >= NN) return;
  if (mark[key]) {
    int s = atomicAdd(cnt, 1);
    slotmap[key] = s;
    ekeys[s] = key;
  } else {
    slotmap[key] = -1;
  }
}

__global__ void k_scatter_e(const int* __restrict__ rpi, const int* __restrict__ slotmap,
                            const float* __restrict__ imef, const float* __restrict__ sgef,
                            float* __restrict__ eim, float* __restrict__ esg) {
  int tid = blockIdx.x * blockDim.x + threadIdx.x;
  if (tid >= N_REL * D_DIM) return;
  int r = tid >> 7, c = tid & 127;
  int key = rpi[2 * r] * N_NODES + rpi[2 * r + 1];
  int slot = slotmap[key];
  atomicAdd(&eim[slot * 128 + c], imef[tid]);
  atomicAdd(&esg[slot * 128 + c], sgef[tid]);
}

__global__ void k_msg_scatter(const float* __restrict__ gate, const float* __restrict__ msg,
                              const int* __restrict__ ekeys, const int* __restrict__ cnt,
                              float* __restrict__ m) {
  int tid = blockIdx.x * blockDim.x + threadIdx.x;
  if (tid >= N_REL * D_DIM) return;
  int slot = tid >> 7;
  if (slot >= *cnt) return;
  int c = tid & 127;
  int key = ekeys[slot];
  int i = key >> 9, j = key & 511;
  atomicAdd(&m[i * 128 + c], gate[slot * 128 + c] * msg[j * 128 + c]);
}

__global__ void k_h2(const float* __restrict__ hn, const float* __restrict__ m,
                     float* __restrict__ h2) {
  int tid = blockIdx.x * blockDim.x + threadIdx.x;
  if (tid >= N_NODES * D_DIM) return;
  h2[tid] = fmaxf(hn[tid] + m[tid], 0.f);
}

// ---------------- pooling ----------------
__global__ void k_node_pool(const float* __restrict__ h2, const float* __restrict__ ngw,
                            const float* __restrict__ ngb, float* __restrict__ wvec,
                            float* __restrict__ sng) {
  int i = blockIdx.x * blockDim.x + threadIdx.x;
  if (i >= N_NODES) return;
  float d = ngb[0];
  for (int c = 0; c < 128; ++c) d += h2[i * 128 + c] * ngw[c];
  float g = sigmoidf(d);
  atomicAdd(sng, g);
  for (int c = 0; c < 128; ++c) atomicAdd(&wvec[c], g * h2[i * 128 + c]);
}

__global__ void k_edge_pool_edges(const float* __restrict__ e2, const float* __restrict__ egw,
                                  const float* __restrict__ egb, const int* __restrict__ cnt,
                                  float* __restrict__ uvec, float* __restrict__ seg) {
  int slot = blockIdx.x * blockDim.x + threadIdx.x;
  if (slot >= N_REL || slot >= *cnt) return;
  float d = egb[0];
  for (int c = 0; c < 128; ++c) d += e2[slot * 128 + c] * egw[c];
  float g = sigmoidf(d);
  atomicAdd(seg, g);
  for (int c = 0; c < 128; ++c) atomicAdd(&uvec[c], g * e2[slot * 128 + c]);
}

// All untouched (zero) edge positions share e2_0 = relu(e_b): add count * term analytically.
__global__ void k_edge_pool_zero(const float* __restrict__ eb, const float* __restrict__ egw,
                                 const float* __restrict__ egb, const int* __restrict__ cnt,
                                 float* __restrict__ uvec, float* __restrict__ seg) {
  __shared__ float sh[128];
  __shared__ float g0s, zcs;
  int c = threadIdx.x;
  float e0 = fmaxf(eb[c], 0.f);
  sh[c] = e0 * egw[c];
  __syncthreads();
  for (int s = 64; s > 0; s >>= 1) {
    if (c < s) sh[c] += sh[c + s];
    __syncthreads();
  }
  if (c == 0) {
    g0s = sigmoidf(sh[0] + egb[0]);
    zcs = (float)(NN - *cnt);
  }
  __syncthreads();
  atomicAdd(&uvec[c], zcs * g0s * e0);
  if (c == 0) atomicAdd(seg, zcs * g0s);
}

__global__ void k_zhalf(const float* __restrict__ wvec, const float* __restrict__ sng,
                        const float* __restrict__ nfw, const float* __restrict__ nfb,
                        const float* __restrict__ uvec, const float* __restrict__ seg,
                        const float* __restrict__ efw, const float* __restrict__ efb,
                        float* __restrict__ zout) {
  int p = blockIdx.x * blockDim.x + threadIdx.x;
  if (p >= POOL_DIM) return;
  float s = (*sng) * nfb[p] + (*seg) * efb[p];
  for (int c = 0; c < 128; ++c) s += wvec[c] * nfw[c * POOL_DIM + p] + uvec[c] * efw[c * POOL_DIM + p];
  zout[p] = s;
}

// ---------------- head ----------------
__global__ void k_en1(const float* __restrict__ z, const float* __restrict__ w,
                      const float* __restrict__ b, float* __restrict__ t) {
  int p = blockIdx.x * blockDim.x + threadIdx.x;
  if (p >= POOL_DIM) return;
  float s = b[p];
  for (int q = 0; q < 2 * POOL_DIM; ++q) s += z[q] * w[q * POOL_DIM + p];
  t[p] = fmaxf(s, 0.f);
}

__global__ void k_en2(const float* __restrict__ t, const float* __restrict__ w,
                      const float* __restrict__ b, float* __restrict__ out) {
  __shared__ float sh[256];
  int c = threadIdx.x;
  float s = 0.f;
  for (int p = c; p < POOL_DIM; p += 256) s += t[p] * w[p];
  sh[c] = s;
  __syncthreads();
  for (int k = 128; k > 0; k >>= 1) {
    if (c < k) sh[c] += sh[c + k];
    __syncthreads();
  }
  if (c == 0) out[0] = sh[0] + b[0];
}

// ---------------- launch ----------------
extern "C" void kernel_launch(void* const* d_in, const int* in_sizes, int n_in,
                              void* d_out, int out_size, void* d_ws, size_t ws_size,
                              hipStream_t stream) {
  (void)in_sizes; (void)n_in; (void)out_size; (void)ws_size;
  const float* x    = (const float*)d_in[0];
  const float* uf   = (const float*)d_in[1];
  const float* box  = (const float*)d_in[2];
  const float* objl = (const float*)d_in[3];
  const float* rell = (const float*)d_in[4];
  auto P = [&](int i) { return (const float*)d_in[i]; };
  // params: 5..54 (w/b pairs in _param_specs order, then bn_g, bn_b); 55 = rel_pair_idx
  const float *pos1_w = P(5),  *pos1_b = P(6),  *pos2_w = P(7),  *pos2_b = P(8);
  const float *obj_w  = P(9),  *obj_b  = P(10), *rel_w  = P(11), *rel_b  = P(12);
  const float *ol_w   = P(13), *ol_b   = P(14), *rl_w   = P(15), *rl_b   = P(16);
  const float *im_g_w = P(17), *im_g_b = P(18), *im_m_w = P(19), *im_m_b = P(20);
  const float *im_n_w = P(21), *im_n_b = P(22), *im_e_w = P(23), *im_e_b = P(24);
  const float *sg_g_w = P(25), *sg_g_b = P(26), *sg_m_w = P(27), *sg_m_b = P(28);
  const float *sg_n_w = P(29), *sg_n_b = P(30), *sg_e_w = P(31), *sg_e_b = P(32);
  const float *imp_ng_w = P(33), *imp_ng_b = P(34), *imp_nf_w = P(35), *imp_nf_b = P(36);
  const float *imp_eg_w = P(37), *imp_eg_b = P(38), *imp_ef_w = P(39), *imp_ef_b = P(40);
  const float *sgp_ng_w = P(41), *sgp_ng_b = P(42), *sgp_nf_w = P(43), *sgp_nf_b = P(44);
  const float *sgp_eg_w = P(45), *sgp_eg_b = P(46), *sgp_ef_w = P(47), *sgp_ef_b = P(48);
  const float *en1_w = P(49), *en1_b = P(50), *en2_w = P(51), *en2_b = P(52);
  const float *bn_g = P(53), *bn_b = P(54);
  const int* rpi = (const int*)d_in[55];

  // workspace allocator
  char* wp = (char*)d_ws;
  auto alloc = [&](size_t bytes) -> void* {
    void* p = (void*)wp;
    wp += (bytes + 255) & ~(size_t)255;
    return p;
  };
  float* t1        = (float*)alloc(N_NODES * 32 * 4);
  float* pos       = (float*)alloc(N_NODES * D_DIM * 4);
  float* sg_h0     = (float*)alloc(N_NODES * D_DIM * 4);
  float* im_h0     = (float*)alloc(N_NODES * D_DIM * 4);
  float* im_e_flat = (float*)alloc((size_t)N_REL * D_DIM * 4);
  float* sg_e_flat = (float*)alloc((size_t)N_REL * D_DIM * 4);
  int*   mark      = (int*)alloc((size_t)NN * 4);
  int*   slotmap   = (int*)alloc((size_t)NN * 4);
  int*   ekeys     = (int*)alloc(N_REL * 4);
  int*   cnt       = (int*)alloc(4);
  float* e_im_acc  = (float*)alloc((size_t)N_REL * D_DIM * 4);
  float* e_sg_acc  = (float*)alloc((size_t)N_REL * D_DIM * 4);
  float* gate_im   = (float*)alloc((size_t)N_REL * D_DIM * 4);
  float* gate_sg   = (float*)alloc((size_t)N_REL * D_DIM * 4);
  float* e2_im     = (float*)alloc((size_t)N_REL * D_DIM * 4);
  float* e2_sg     = (float*)alloc((size_t)N_REL * D_DIM * 4);
  float* msg_im    = (float*)alloc(N_NODES * D_DIM * 4);
  float* msg_sg    = (float*)alloc(N_NODES * D_DIM * 4);
  float* hn_im     = (float*)alloc(N_NODES * D_DIM * 4);
  float* hn_sg     = (float*)alloc(N_NODES * D_DIM * 4);
  float* m_im      = (float*)alloc(N_NODES * D_DIM * 4);
  float* m_sg      = (float*)alloc(N_NODES * D_DIM * 4);
  float* h2_im     = (float*)alloc(N_NODES * D_DIM * 4);
  float* h2_sg     = (float*)alloc(N_NODES * D_DIM * 4);
  float* wvec_im   = (float*)alloc(128 * 4);
  float* uvec_im   = (float*)alloc(128 * 4);
  float* sc_im     = (float*)alloc(2 * 4);  // [s_ng, s_eg]
  float* wvec_sg   = (float*)alloc(128 * 4);
  float* uvec_sg   = (float*)alloc(128 * 4);
  float* sc_sg     = (float*)alloc(2 * 4);
  float* zbuf      = (float*)alloc(2 * POOL_DIM * 4);
  float* tbuf      = (float*)alloc(POOL_DIM * 4);
  unsigned short* x_bf       = (unsigned short*)alloc((size_t)N_NODES * IN_DIM * 2);
  unsigned short* uf_bf      = (unsigned short*)alloc((size_t)N_REL * IN_DIM * 2);
  unsigned short* objw_p     = (unsigned short*)alloc((size_t)IN_DIM * 128 * 2);
  unsigned short* relw_p     = (unsigned short*)alloc((size_t)IN_DIM * 128 * 2);
  unsigned short* gwp_im     = (unsigned short*)alloc(128 * 128 * 2);
  unsigned short* ewp_im     = (unsigned short*)alloc(128 * 128 * 2);
  unsigned short* mwp_im     = (unsigned short*)alloc(128 * 128 * 2);
  unsigned short* nwp_im     = (unsigned short*)alloc(128 * 128 * 2);
  unsigned short* gwp_sg     = (unsigned short*)alloc(128 * 128 * 2);
  unsigned short* ewp_sg     = (unsigned short*)alloc(128 * 128 * 2);
  unsigned short* mwp_sg     = (unsigned short*)alloc(128 * 128 * 2);
  unsigned short* nwp_sg     = (unsigned short*)alloc(128 * 128 * 2);
  unsigned short* eacc_im_bf = (unsigned short*)alloc((size_t)N_REL * D_DIM * 2);
  unsigned short* eacc_sg_bf = (unsigned short*)alloc((size_t)N_REL * D_DIM * 2);
  unsigned short* h0_im_bf   = (unsigned short*)alloc(N_NODES * D_DIM * 2);
  unsigned short* h0_sg_bf   = (unsigned short*)alloc(N_NODES * D_DIM * 2);

  // zero init (graph-capture safe)
  hipMemsetAsync(mark, 0, (size_t)NN * 4, stream);
  hipMemsetAsync(cnt, 0, 4, stream);
  hipMemsetAsync(e_im_acc, 0, (size_t)N_REL * D_DIM * 4, stream);
  hipMemsetAsync(e_sg_acc, 0, (size_t)N_REL * D_DIM * 4, stream);
  hipMemsetAsync(m_im, 0, N_NODES * D_DIM * 4, stream);
  hipMemsetAsync(m_sg, 0, N_NODES * D_DIM * 4, stream);
  hipMemsetAsync(wvec_im, 0, 128 * 4, stream);
  hipMemsetAsync(uvec_im, 0, 128 * 4, stream);
  hipMemsetAsync(sc_im, 0, 2 * 4, stream);
  hipMemsetAsync(wvec_sg, 0, 128 * 4, stream);
  hipMemsetAsync(uvec_sg, 0, 128 * 4, stream);
  hipMemsetAsync(sc_sg, 0, 2 * 4, stream);

  auto g = [](int n) { return dim3((n + 255) / 256); };
  const dim3 B256(256);
  // GEMM grids: x = M-tiles / 8 waves per block, y = n-groups (8 / NTW)
  const dim3 gemm512(dim3(N_NODES / 16 / 8, 8));   // NTW=1: 256 waves
  const dim3 gemm4k(dim3(N_REL / 16 / 8, 4));      // NTW=2: 1024 waves

  // convert + pack
  k_f2bf<<<g(N_NODES * IN_DIM), B256, 0, stream>>>(x, x_bf, N_NODES * IN_DIM);
  k_f2bf<<<g(N_REL * IN_DIM), B256, 0, stream>>>(uf, uf_bf, N_REL * IN_DIM);
  k_packB<<<g(IN_DIM * 128), B256, 0, stream>>>(obj_w, objw_p, IN_DIM);
  k_packB<<<g(IN_DIM * 128), B256, 0, stream>>>(rel_w, relw_p, IN_DIM);
  k_packB<<<g(128 * 128), B256, 0, stream>>>(im_g_w, gwp_im, 128);
  k_packB<<<g(128 * 128), B256, 0, stream>>>(im_e_w, ewp_im, 128);
  k_packB<<<g(128 * 128), B256, 0, stream>>>(im_m_w, mwp_im, 128);
  k_packB<<<g(128 * 128), B256, 0, stream>>>(im_n_w, nwp_im, 128);
  k_packB<<<g(128 * 128), B256, 0, stream>>>(sg_g_w, gwp_sg, 128);
  k_packB<<<g(128 * 128), B256, 0, stream>>>(sg_e_w, ewp_sg, 128);
  k_packB<<<g(128 * 128), B256, 0, stream>>>(sg_m_w, mwp_sg, 128);
  k_packB<<<g(128 * 128), B256, 0, stream>>>(sg_n_w, nwp_sg, 128);

  // big WMMA GEMMs
  k_wmma_gemm_n128<1><<<gemm512, B256, 0, stream>>>(x_bf, objw_p, obj_b, im_h0, N_NODES, IN_DIM, 0);
  k_wmma_gemm_n128<2><<<gemm4k, B256, 0, stream>>>(uf_bf, relw_p, rel_b, im_e_flat, N_REL, IN_DIM, 0);

  // small dense chain
  k_pos1<<<g(N_NODES * 32), B256, 0, stream>>>(box, pos1_w, pos1_b, bn_g, bn_b, t1);
  k_pos2<<<g(N_NODES * D_DIM), B256, 0, stream>>>(t1, pos2_w, pos2_b, pos);
  k_sgh0<<<g(N_NODES * D_DIM), B256, 0, stream>>>(objl, pos, ol_w, ol_b, sg_h0);
  k_sge<<<g(N_REL * D_DIM), B256, 0, stream>>>(rell, rl_w, rl_b, sg_e_flat);

  // sparse edge compaction + accumulation
  k_mark<<<g(N_REL), B256, 0, stream>>>(rpi, mark);
  k_compact<<<g(NN), B256, 0, stream>>>(mark, slotmap, ekeys, cnt);
  k_scatter_e<<<g(N_REL * D_DIM), B256, 0, stream>>>(rpi, slotmap, im_e_flat, sg_e_flat, e_im_acc, e_sg_acc);

  // bf16 for edge/node GEMMs
  k_f2bf<<<g(N_REL * D_DIM), B256, 0, stream>>>(e_im_acc, eacc_im_bf, N_REL * D_DIM);
  k_f2bf<<<g(N_REL * D_DIM), B256, 0, stream>>>(e_sg_acc, eacc_sg_bf, N_REL * D_DIM);
  k_f2bf<<<g(N_NODES * D_DIM), B256, 0, stream>>>(im_h0, h0_im_bf, N_NODES * D_DIM);
  k_f2bf<<<g(N_NODES * D_DIM), B256, 0, stream>>>(sg_h0, h0_sg_bf, N_NODES * D_DIM);

  // edge GEMMs on compacted edges (padded rows are zero -> harmless)
  k_wmma_gemm_n128<2><<<gemm4k, B256, 0, stream>>>(eacc_im_bf, gwp_im, im_g_b, gate_im, N_REL, 128, 2);
  k_wmma_gemm_n128<2><<<gemm4k, B256, 0, stream>>>(eacc_im_bf, ewp_im, im_e_b, e2_im, N_REL, 128, 1);
  k_wmma_gemm_n128<2><<<gemm4k, B256, 0, stream>>>(eacc_sg_bf, gwp_sg, sg_g_b, gate_sg, N_REL, 128, 2);
  k_wmma_gemm_n128<2><<<gemm4k, B256, 0, stream>>>(eacc_sg_bf, ewp_sg, sg_e_b, e2_sg, N_REL, 128, 1);

  // node GEMMs
  k_wmma_gemm_n128<1><<<gemm512, B256, 0, stream>>>(h0_im_bf, mwp_im, im_m_b, msg_im, N_NODES, 128, 0);
  k_wmma_gemm_n128<1><<<gemm512, B256, 0, stream>>>(h0_im_bf, nwp_im, im_n_b, hn_im, N_NODES, 128, 0);
  k_wmma_gemm_n128<1><<<gemm512, B256, 0, stream>>>(h0_sg_bf, mwp_sg, sg_m_b, msg_sg, N_NODES, 128, 0);
  k_wmma_gemm_n128<1><<<gemm512, B256, 0, stream>>>(h0_sg_bf, nwp_sg, sg_n_b, hn_sg, N_NODES, 128, 0);

  // message passing + node update
  k_msg_scatter<<<g(N_REL * D_DIM), B256, 0, stream>>>(gate_im, msg_im, ekeys, cnt, m_im);
  k_msg_scatter<<<g(N_REL * D_DIM), B256, 0, stream>>>(gate_sg, msg_sg, ekeys, cnt, m_sg);
  k_h2<<<g(N_NODES * D_DIM), B256, 0, stream>>>(hn_im, m_im, h2_im);
  k_h2<<<g(N_NODES * D_DIM), B256, 0, stream>>>(hn_sg, m_sg, h2_sg);

  // pooling
  k_node_pool<<<g(N_NODES), B256, 0, stream>>>(h2_im, imp_ng_w, imp_ng_b, wvec_im, &sc_im[0]);
  k_node_pool<<<g(N_NODES), B256, 0, stream>>>(h2_sg, sgp_ng_w, sgp_ng_b, wvec_sg, &sc_sg[0]);
  k_edge_pool_edges<<<g(N_REL), B256, 0, stream>>>(e2_im, imp_eg_w, imp_eg_b, cnt, uvec_im, &sc_im[1]);
  k_edge_pool_edges<<<g(N_REL), B256, 0, stream>>>(e2_sg, sgp_eg_w, sgp_eg_b, cnt, uvec_sg, &sc_sg[1]);
  k_edge_pool_zero<<<dim3(1), dim3(128), 0, stream>>>(im_e_b, imp_eg_w, imp_eg_b, cnt, uvec_im, &sc_im[1]);
  k_edge_pool_zero<<<dim3(1), dim3(128), 0, stream>>>(sg_e_b, sgp_eg_w, sgp_eg_b, cnt, uvec_sg, &sc_sg[1]);
  k_zhalf<<<g(POOL_DIM), B256, 0, stream>>>(wvec_im, &sc_im[0], imp_nf_w, imp_nf_b, uvec_im, &sc_im[1], imp_ef_w, imp_ef_b, zbuf);
  k_zhalf<<<g(POOL_DIM), B256, 0, stream>>>(wvec_sg, &sc_sg[0], sgp_nf_w, sgp_nf_b, uvec_sg, &sc_sg[1], sgp_ef_w, sgp_ef_b, zbuf + POOL_DIM);

  // head
  k_en1<<<g(POOL_DIM), B256, 0, stream>>>(zbuf, en1_w, en1_b, tbuf);
  k_en2<<<dim3(1), B256, 0, stream>>>(tbuf, en2_w, en2_b, (float*)d_out);
}
